// QLSTM_65481071406473
// MI455X (gfx1250) — matrix operand order repaired and checked by reading, not compiled, not run on hardware
//
#include <hip/hip_runtime.h>
#include <hip/hip_bf16.h>

typedef __attribute__((ext_vector_type(16))) _Float16 v16h;
typedef __attribute__((ext_vector_type(8)))  float    v8f;

#define T_STEPS 512
#define B_ROWS  512
#define K_DIM   128
#define N_COLS  32          // 4 gates x 8 qubits
#define M_ROWS  (T_STEPS * B_ROWS)
#define M_TILES (M_ROWS / 16)

// ---------------------------------------------------------------------------
// Kernel 1: Zx[t*B+b, g*8+q] = x[t,b,:] @ Wg[:128, q] + bg[q]
// One wave per 16-row M-tile; both 16-col N-tiles per wave.
// v_wmma_f32_16x16x32_f16, K-loop of 4.
// ---------------------------------------------------------------------------
__global__ __launch_bounds__(256) void qlstm_xproj(
    const float* __restrict__ x,
    const float* __restrict__ Wf, const float* __restrict__ Wi,
    const float* __restrict__ Wu, const float* __restrict__ Wo,
    const float* __restrict__ bf, const float* __restrict__ bi,
    const float* __restrict__ bu, const float* __restrict__ bo,
    float* __restrict__ Zx)
{
    const int lane    = threadIdx.x & 31;
    const int wave    = threadIdx.x >> 5;
    const int mtile   = blockIdx.x * 8 + wave;          // 0 .. 16383
    const int halfsel = lane >> 4;                      // 0|1 (lane half)
    const int l16     = lane & 15;

    // ---- B fragments (32x16 f16, K=(half)*8+e (+8 for e>=8), N=l16), both N-tiles ----
    const float* W0 = (l16 < 8) ? Wf : Wi;  const int q0 = l16 & 7;  // cols 0..15
    const float* W1 = (l16 < 8) ? Wu : Wo;  const int q1 = l16 & 7;  // cols 16..31
    const float bias0 = ((l16 < 8) ? bf : bi)[q0];
    const float bias1 = ((l16 < 8) ? bu : bo)[q1];

    v16h Bf0[4], Bf1[4];
    #pragma unroll
    for (int kt = 0; kt < 4; ++kt) {
        const int kbase = kt * 32 + halfsel * 8;
        v16h b0, b1;
        #pragma unroll
        for (int e = 0; e < 16; ++e) {
            b0[e] = (_Float16)W0[(kbase + e + (e & 8)) * 8 + q0];
            b1[e] = (_Float16)W1[(kbase + e + (e & 8)) * 8 + q1];
        }
        Bf0[kt] = b0; Bf1[kt] = b1;
    }

    // ---- A row pointer: row m = mtile*16 + l16, per-lane K runs ----
    const float* arow = x + (size_t)(mtile * 16 + l16) * K_DIM + halfsel * 8;

    v8f c0 = {}; v8f c1 = {};
    #pragma unroll
    for (int kt = 0; kt < 4; ++kt) {
        const float4* p1 = reinterpret_cast<const float4*>(arow + kt * 32);
        const float4* p2 = reinterpret_cast<const float4*>(arow + kt * 32 + 16);
        float4 a0 = p1[0], a1 = p1[1];      // K = kt*32 + h8 + 0..7
        float4 a2 = p2[0], a3 = p2[1];      // K = kt*32 + 16 + h8 + 0..7
        v16h a;
        a[0]  = (_Float16)a0.x; a[1]  = (_Float16)a0.y;
        a[2]  = (_Float16)a0.z; a[3]  = (_Float16)a0.w;
        a[4]  = (_Float16)a1.x; a[5]  = (_Float16)a1.y;
        a[6]  = (_Float16)a1.z; a[7]  = (_Float16)a1.w;
        a[8]  = (_Float16)a2.x; a[9]  = (_Float16)a2.y;
        a[10] = (_Float16)a2.z; a[11] = (_Float16)a2.w;
        a[12] = (_Float16)a3.x; a[13] = (_Float16)a3.y;
        a[14] = (_Float16)a3.z; a[15] = (_Float16)a3.w;

        c0 = __builtin_amdgcn_wmma_f32_16x16x32_f16(false, a, false, Bf0[kt],
                                                    (short)0, c0, false, false);
        c1 = __builtin_amdgcn_wmma_f32_16x16x32_f16(false, a, false, Bf1[kt],
                                                    (short)0, c1, false, false);
    }

    // ---- store C (M=(half)*8+v, N=l16) + bias ----
    float* zbase = Zx + (size_t)mtile * 16 * N_COLS;
    #pragma unroll
    for (int v = 0; v < 8; ++v) {
        const int r = halfsel * 8 + v;
        zbase[(size_t)r * N_COLS + l16]      = c0[v] + bias0;
        zbase[(size_t)r * N_COLS + 16 + l16] = c1[v] + bias1;
    }
}

// ---------------------------------------------------------------------------
// Kernel 2: sequential scan. One lane per batch row.
// qlayer collapses to prefix/suffix products of cos(z + theta).
// ---------------------------------------------------------------------------
__device__ __forceinline__ float fsig(float v) {
    return __builtin_amdgcn_rcpf(1.0f + __expf(-v));
}
__device__ __forceinline__ float ftanh(float v) {
    float e = __expf(-2.0f * v);
    return (1.0f - e) * __builtin_amdgcn_rcpf(1.0f + e);
}

__global__ __launch_bounds__(256, 1) void qlstm_rec(
    const float* __restrict__ Zx,
    const float* __restrict__ Wf, const float* __restrict__ Wi,
    const float* __restrict__ Wu, const float* __restrict__ Wo,
    const float* __restrict__ thf, const float* __restrict__ thi,
    const float* __restrict__ thu, const float* __restrict__ tho,
    float* __restrict__ out)
{
    __shared__ float Whs[8][N_COLS];   // hidden-part weights, rows 128..135
    __shared__ float ths[N_COLS];

    const int tid = threadIdx.x;
    {
        const int k = tid >> 5, j = tid & 31;
        const int g = j >> 3, q = j & 7;
        const float* W = (g == 0) ? Wf : (g == 1) ? Wi : (g == 2) ? Wu : Wo;
        Whs[k][j] = W[(K_DIM + k) * 8 + q];
        if (k == 0) {
            const float* th = (g == 0) ? thf : (g == 1) ? thi : (g == 2) ? thu : tho;
            ths[j] = th[q];
        }
    }
    __syncthreads();

    const int row = blockIdx.x * 256 + tid;

    float th_r[N_COLS];
    #pragma unroll
    for (int j = 0; j < N_COLS; ++j) th_r[j] = ths[j];

    float hx[8], cx[8];
    #pragma unroll
    for (int q = 0; q < 8; ++q) { hx[q] = 0.0f; cx[q] = 0.0f; }

    // prefetched x-projection for the current step
    float z[N_COLS];
    {
        const float4* zp = reinterpret_cast<const float4*>(Zx + (size_t)row * N_COLS);
        #pragma unroll
        for (int j4 = 0; j4 < 8; ++j4) {
            float4 v = zp[j4];
            z[4*j4+0] = v.x; z[4*j4+1] = v.y; z[4*j4+2] = v.z; z[4*j4+3] = v.w;
        }
    }

    for (int t = 0; t < T_STEPS; ++t) {
        // z_total = Zx (with bias) + hx @ Wh
        float zz[N_COLS];
        #pragma unroll
        for (int j = 0; j < N_COLS; ++j) zz[j] = z[j];
        #pragma unroll
        for (int k = 0; k < 8; ++k) {
            const float h = hx[k];
            #pragma unroll
            for (int j = 0; j < N_COLS; ++j) zz[j] = fmaf(h, Whs[k][j], zz[j]);
        }

        // prefetch next step's projection (overlaps the transcendental work)
        if (t + 1 < T_STEPS) {
            const float4* zp = reinterpret_cast<const float4*>(
                Zx + ((size_t)(t + 1) * B_ROWS + row) * N_COLS);
            #pragma unroll
            for (int j4 = 0; j4 < 8; ++j4) {
                float4 v = zp[j4];
                z[4*j4+0] = v.x; z[4*j4+1] = v.y; z[4*j4+2] = v.z; z[4*j4+3] = v.w;
            }
        }

        // quantum layer: out[q] = prod_{w in S_q} cos(z_w + theta_w)
        float cv[N_COLS];
        #pragma unroll
        for (int j = 0; j < N_COLS; ++j) cv[j] = __cosf(zz[j] + th_r[j]);

        float go[N_COLS];
        #pragma unroll
        for (int g = 0; g < 4; ++g) {
            const float* c = cv + g * 8;
            float pref = c[0];
            #pragma unroll
            for (int q = 1; q < 8; ++q) { pref *= c[q]; go[g*8 + q] = pref; }
            float suf = c[7];
            #pragma unroll
            for (int q = 6; q >= 1; --q) suf *= c[q];
            go[g*8 + 0] = suf;
        }

        // LSTM combine
        #pragma unroll
        for (int q = 0; q < 8; ++q) {
            const float fg = fsig(go[q]);
            const float ig = fsig(go[8 + q]);
            const float ug = ftanh(go[16 + q]);
            const float og = fsig(go[24 + q]);
            const float c2 = fmaf(fg, cx[q], ig * ug);
            cx[q] = c2;
            hx[q] = og * ftanh(c2);
        }

        float4* op = reinterpret_cast<float4*>(out + ((size_t)t * B_ROWS + row) * 8);
        op[0] = make_float4(hx[0], hx[1], hx[2], hx[3]);
        op[1] = make_float4(hx[4], hx[5], hx[6], hx[7]);
    }

    // trailing hx, cx outputs
    float* hp = out + (size_t)T_STEPS * B_ROWS * 8 + (size_t)row * 8;
    float* cp = hp + (size_t)B_ROWS * 8;
    reinterpret_cast<float4*>(hp)[0] = make_float4(hx[0], hx[1], hx[2], hx[3]);
    reinterpret_cast<float4*>(hp)[1] = make_float4(hx[4], hx[5], hx[6], hx[7]);
    reinterpret_cast<float4*>(cp)[0] = make_float4(cx[0], cx[1], cx[2], cx[3]);
    reinterpret_cast<float4*>(cp)[1] = make_float4(cx[4], cx[5], cx[6], cx[7]);
}

// ---------------------------------------------------------------------------
extern "C" void kernel_launch(void* const* d_in, const int* in_sizes, int n_in,
                              void* d_out, int out_size, void* d_ws, size_t ws_size,
                              hipStream_t stream) {
    (void)in_sizes; (void)n_in; (void)out_size; (void)ws_size;
    const float* x   = (const float*)d_in[0];
    const float* Wf  = (const float*)d_in[1];
    const float* bf  = (const float*)d_in[2];
    const float* Wi  = (const float*)d_in[3];
    const float* bi  = (const float*)d_in[4];
    const float* Wu  = (const float*)d_in[5];
    const float* bu  = (const float*)d_in[6];
    const float* Wo  = (const float*)d_in[7];
    const float* bo  = (const float*)d_in[8];
    const float* thf = (const float*)d_in[9];
    const float* thi = (const float*)d_in[10];
    const float* thu = (const float*)d_in[11];
    const float* tho = (const float*)d_in[12];

    float* Zx  = (float*)d_ws;        // M_ROWS * 32 floats = 33.5 MB (L2-resident)
    float* out = (float*)d_out;

    // 16384 M-tiles, 8 waves (one tile each) per 256-thread block
    qlstm_xproj<<<M_TILES / 8, 256, 0, stream>>>(x, Wf, Wi, Wu, Wo,
                                                 bf, bi, bu, bo, Zx);
    // 512 rows, one lane each
    qlstm_rec<<<B_ROWS / 256, 256, 0, stream>>>(Zx, Wf, Wi, Wu, Wo,
                                                thf, thi, thu, tho, out);
}
